// MultiAttentionHead_90598040142283
// MI455X (gfx1250) — matrix operand order repaired and checked by reading, not compiled, not run on hardware
//
#include <hip/hip_runtime.h>

typedef __attribute__((ext_vector_type(16))) _Float16 v16h;
typedef __attribute__((ext_vector_type(8)))  _Float16 v8h;
typedef __attribute__((ext_vector_type(8)))  float    v8f;

#define STR 136            // padded LDS row stride in halves (272 B -> bank-rotating)
#define T_CTX 128
#define D_EMB 128

union F16Frag { v16h v; v8h h[2]; };

// A-fragment (16x32 f16): lane<16 holds row=lane, K {0..7,16..23}; lane>=16 holds K {8..15,24..31}
__device__ __forceinline__ v16h fragA(const _Float16* base, int stride) {
  const int lane = threadIdx.x & 31;
  const _Float16* p = base + (lane & 15) * stride + (lane >> 4) * 8;
  F16Frag f;
  f.h[0] = *(const v8h*)(p);
  f.h[1] = *(const v8h*)(p + 16);
  return f.v;
}

// B-fragment (32x16 f16), source stored column-major (row n = output col, contiguous in K):
// lane<16 holds col=lane, K {0..7,16..23}; lane>=16 holds col=lane-16, K {8..15,24..31}
__device__ __forceinline__ v16h fragB(const _Float16* base, int stride) {
  const int lane = threadIdx.x & 31;
  const _Float16* p = base + (lane & 15) * stride + (lane >> 4) * 8;
  F16Frag f;
  f.h[0] = *(const v8h*)(p);
  f.h[1] = *(const v8h*)(p + 16);
  return f.v;
}

__device__ __forceinline__ v8f wmma16(v16h a, v16h b, v8f c) {
  return __builtin_amdgcn_wmma_f32_16x16x32_f16(false, a, false, b, (short)0, c, false, false);
}

// ---- weight prep: fp32 -> fp16, column-major [n][d] --------------------------
__global__ void prep_weights(const float* __restrict__ Wq, const float* __restrict__ Wk,
                             const float* __restrict__ Wv, const float* __restrict__ Wp,
                             _Float16* __restrict__ wqkv, _Float16* __restrict__ wpt) {
  int i = blockIdx.x * 256 + threadIdx.x;
  if (i < 3 * 128 * 128) {
    int m = i / 16384;          // 0=Q 1=K 2=V
    int r = i - m * 16384;
    int n = r >> 7;             // output column (h*32 + k)
    int d = r & 127;
    const float* W = (m == 0) ? Wq : (m == 1) ? Wk : Wv;
    int h = n >> 5, k = n & 31;
    wqkv[i] = (_Float16)W[h * 4096 + d * 32 + k];   // W[h][d][k]
  } else {
    int r = i - 3 * 128 * 128;
    int n = r >> 7, d = r & 127;
    wpt[n * 128 + d] = (_Float16)Wp[d * 128 + n];   // transpose Wp
  }
}

// ---- fused MHA: one block per batch element ---------------------------------
__global__ __launch_bounds__(256)
void mha_kernel(const float* __restrict__ x,
                const float* __restrict__ bq, const float* __restrict__ bk,
                const float* __restrict__ bv, const float* __restrict__ bp,
                const _Float16* __restrict__ wqkv, const _Float16* __restrict__ wpt,
                float* __restrict__ out) {
  __shared__ _Float16 sQ[T_CTX * STR];      // Q [t][d], later overwritten by O
  __shared__ _Float16 sK[T_CTX * STR];      // K [t][d]
  __shared__ _Float16 sV[4 * 32 * STR];     // V transposed per head: [h][hd][t]
  __shared__ _Float16 sP[8 * 16 * STR];     // per-wave softmax scratch [16][t]

  const int tid  = threadIdx.x;
  const int lane = tid & 31;
  const int w    = tid >> 5;     // wave 0..7
  const int col  = lane & 15;
  const int rh   = lane >> 4;    // 0 or 1 (row half of C fragment)
  const int bidx = blockIdx.x;
  const float* xb = x + (size_t)bidx * (T_CTX * D_EMB);

  // warm the weight caches
  if (tid < 128)       __builtin_prefetch(&wqkv[tid * 768], 0, 1);
  else if (tid < 160)  __builtin_prefetch(&wpt[(tid - 128) * 512], 0, 1);

  // ================= Phase 1: Q,K,V projections ==============================
  // wave w owns query row-tile m0 = 16*w ; x A-fragments loaded once, f32->f16
  const int m0 = w * 16;
  v16h xa[4];
  {
    const float* px = xb + (size_t)m0 * D_EMB + (lane & 15) * D_EMB + rh * 8;
#pragma unroll
    for (int kk = 0; kk < 4; ++kk) {
      const float* p = px + kk * 32;
      float4 f0 = *(const float4*)(p);
      float4 f1 = *(const float4*)(p + 4);
      float4 f2 = *(const float4*)(p + 16);
      float4 f3 = *(const float4*)(p + 20);
      v16h a;
      a[0]=(_Float16)f0.x; a[1]=(_Float16)f0.y; a[2]=(_Float16)f0.z; a[3]=(_Float16)f0.w;
      a[4]=(_Float16)f1.x; a[5]=(_Float16)f1.y; a[6]=(_Float16)f1.z; a[7]=(_Float16)f1.w;
      a[8]=(_Float16)f2.x; a[9]=(_Float16)f2.y; a[10]=(_Float16)f2.z; a[11]=(_Float16)f2.w;
      a[12]=(_Float16)f3.x; a[13]=(_Float16)f3.y; a[14]=(_Float16)f3.z; a[15]=(_Float16)f3.w;
      xa[kk] = a;
    }
  }
#pragma unroll
  for (int mat = 0; mat < 3; ++mat) {
    const float* bias = (mat == 0) ? bq : (mat == 1) ? bk : bv;
#pragma unroll
    for (int nt = 0; nt < 8; ++nt) {
      const _Float16* wb = wqkv + (mat * 128 + nt * 16) * 128;
      // issue all 8 b128 weight loads as one clause, then the 4-WMMA chain
      v16h bf[4];
#pragma unroll
      for (int kk = 0; kk < 4; ++kk) bf[kk] = fragB(wb + kk * 32, 128);
      v8f c = {};
#pragma unroll
      for (int kk = 0; kk < 4; ++kk) c = wmma16(xa[kk], bf[kk], c);
      const float bia = bias[nt * 16 + col];
      if (mat == 0) {
        _Float16* dst = sQ + m0 * STR + nt * 16;
#pragma unroll
        for (int g = 0; g < 8; ++g) dst[(g + 8 * rh) * STR + col] = (_Float16)(c[g] + bia);
      } else if (mat == 1) {
        _Float16* dst = sK + m0 * STR + nt * 16;
#pragma unroll
        for (int g = 0; g < 8; ++g) dst[(g + 8 * rh) * STR + col] = (_Float16)(c[g] + bia);
      } else {
        const int h = nt >> 1;
        const int kl = (nt & 1) * 16 + col;          // hd index within head
        _Float16* dst = sV + (h * 32 + kl) * STR + m0;
#pragma unroll
        for (int g = 0; g < 8; ++g) dst[g + 8 * rh] = (_Float16)(c[g] + bia);
      }
    }
  }
  __syncthreads();

  // ================= Phase 2: causal attention per head ======================
  // All loops over score tiles n / K-steps ks are fully unrolled with
  // *wave-uniform* guards (qtile is uniform) -> sc[] stays in registers and
  // dead work beyond the causal frontier is skipped by scalar branches.
  {
    const int h = w >> 1, sub = w & 1;               // 2 waves per head
    const float scale = 0.08838834764831845f;        // CONTEXT_SIZE**-0.5
    _Float16* Pw = sP + w * 16 * STR;
#pragma unroll
    for (int qi = 0; qi < 4; ++qi) {
      const int qtile = sub * 4 + qi;                // wave-uniform
      const int q0 = qtile * 16;
      const v16h aq = fragA(sQ + q0 * STR + h * 32, STR);
      v8f sc[8];
#pragma unroll
      for (int n = 0; n < 8; ++n) {
        if (n <= qtile)                              // live tile: Q.K^T (K = hd = 32)
          sc[n] = wmma16(aq, fragB(sK + (n * 16) * STR + h * 32, STR), v8f{});
      }
      float rs[8];
#pragma unroll
      for (int g = 0; g < 8; ++g) {
        const int q = q0 + g + 8 * rh;
        float mx = -1e30f;
#pragma unroll
        for (int n = 0; n < 8; ++n)
          if (n <= qtile) {
            float v = sc[n][g] * scale;
            if (n * 16 + col > q) v = -1e30f;        // causal mask
            sc[n][g] = v;
            mx = fmaxf(mx, v);
          }
#pragma unroll
        for (int mm = 1; mm < 16; mm <<= 1) mx = fmaxf(mx, __shfl_xor(mx, mm, 32));
        float sum = 0.f;
#pragma unroll
        for (int n = 0; n < 8; ++n)
          if (n <= qtile) {
            float e = __expf(sc[n][g] - mx);         // masked entries -> 0
            sc[n][g] = e;
            sum += e;
          }
#pragma unroll
        for (int mm = 1; mm < 16; mm <<= 1) sum += __shfl_xor(sum, mm, 32);
        rs[g] = 1.0f / sum;
      }
      // relayout P through per-wave LDS scratch (C-frag -> A-frag)
#pragma unroll
      for (int n = 0; n < 8; ++n)
        if (n <= qtile)
#pragma unroll
          for (int g = 0; g < 8; ++g)
            Pw[(g + 8 * rh) * STR + n * 16 + col] = (_Float16)sc[n][g];
      if (!(qtile & 1)) {                            // zero the dangling half-tile
#pragma unroll
        for (int g = 0; g < 8; ++g)
          Pw[(g + 8 * rh) * STR + (qtile + 1) * 16 + col] = (_Float16)0.f;
      }
      const int nk = (qtile + 2) >> 1;               // 32-wide K-steps covering n<=qtile
      v8f o0 = {}, o1 = {};
#pragma unroll
      for (int ks = 0; ks < 4; ++ks)
        if (ks < nk) {                               // uniform scalar branch
          const v16h ap = fragA(Pw + ks * 32, STR);
          o0 = wmma16(ap, fragB(sV + (h * 32 + 0)  * STR + ks * 32, STR), o0);
          o1 = wmma16(ap, fragB(sV + (h * 32 + 16) * STR + ks * 32, STR), o1);
        }
      // O overwrites our own Q slice (rows q0..q0+15, cols h*32..h*32+31)
      _Float16* dst = sQ + q0 * STR + h * 32;
#pragma unroll
      for (int g = 0; g < 8; ++g) {
        dst[(g + 8 * rh) * STR + col]      = (_Float16)(o0[g] * rs[g]);
        dst[(g + 8 * rh) * STR + 16 + col] = (_Float16)(o1[g] * rs[g]);
      }
    }
  }
  __syncthreads();

  // ================= Phase 3: output projection ==============================
  {
    v16h oa[4];
#pragma unroll
    for (int kk = 0; kk < 4; ++kk) oa[kk] = fragA(sQ + m0 * STR + kk * 32, STR);
    float* ob = out + (size_t)bidx * (T_CTX * D_EMB) + (size_t)m0 * D_EMB;
#pragma unroll
    for (int nt = 0; nt < 8; ++nt) {
      const _Float16* wb = wpt + nt * 16 * 128;
      v16h bf[4];
#pragma unroll
      for (int kk = 0; kk < 4; ++kk) bf[kk] = fragB(wb + kk * 32, 128);
      v8f c = {};
#pragma unroll
      for (int kk = 0; kk < 4; ++kk) c = wmma16(oa[kk], bf[kk], c);
      const float bb = bp[nt * 16 + col];
#pragma unroll
      for (int g = 0; g < 8; ++g)
        ob[(g + 8 * rh) * D_EMB + nt * 16 + col] = c[g] + bb;
    }
  }
}

extern "C" void kernel_launch(void* const* d_in, const int* in_sizes, int n_in,
                              void* d_out, int out_size, void* d_ws, size_t ws_size,
                              hipStream_t stream) {
  const float* x  = (const float*)d_in[0];
  const float* Wq = (const float*)d_in[1];
  const float* bq = (const float*)d_in[2];
  const float* Wk = (const float*)d_in[3];
  const float* bk = (const float*)d_in[4];
  const float* Wv = (const float*)d_in[5];
  const float* bv = (const float*)d_in[6];
  const float* Wp = (const float*)d_in[7];
  const float* bp = (const float*)d_in[8];
  float* out = (float*)d_out;

  _Float16* wqkv = (_Float16*)d_ws;            // [3*128][128] f16 column-major
  _Float16* wpt  = wqkv + 384 * 128;           // [128][128] f16 (Wp transposed)

  const int B = in_sizes[0] / (T_CTX * D_EMB); // 4096

  prep_weights<<<(4 * 128 * 128) / 256, 256, 0, stream>>>(Wq, Wk, Wv, Wp, wqkv, wpt);
  mha_kernel<<<B, 256, 0, stream>>>(x, bq, bk, bv, bp, wqkv, wpt, out);
}
